// AlphaWeightedHead_67654324846726
// MI455X (gfx1250) — matrix operand order repaired and checked by reading, not compiled, not run on hardware
//
#include <hip/hip_runtime.h>

// ---------------------------------------------------------------- constants
#define Bn   16
#define Nn   1024
#define Cn   768
#define Hn   12
#define HDn  64
#define D3n  2304   // 3*C

typedef __attribute__((ext_vector_type(16))) __bf16 bf16x16;
typedef __attribute__((ext_vector_type(8)))  __bf16 bf16x8;
typedef __attribute__((ext_vector_type(8)))  float  f32x8;
typedef int v4i_vs __attribute__((vector_size(16)));   // matches builtin param

// ---------------------------------------------------------------- WMMA helpers
__device__ __forceinline__ f32x8 wmma_bf16(bf16x16 a, bf16x16 b, f32x8 c) {
  return __builtin_amdgcn_wmma_f32_16x16x32_bf16(false, a, false, b, (short)0, c,
                                                 false, false);
}

// A-fragment (16x32, bf16, row-major source with element stride `stride`):
// lane l: m = l&15 ; element e -> k = (l>=16?8:0) + (e&7) + (e>=8?16:0)
__device__ __forceinline__ bf16x16 load_a_frag(const __bf16* base, int stride,
                                               int row0, int k0) {
  int l   = threadIdx.x & 31;
  int m   = l & 15;
  int klo = (l & 16) ? 8 : 0;
  const __bf16* p = base + (size_t)(row0 + m) * stride + (k0 + klo);
  bf16x8 lo = *(const bf16x8*)(p);
  bf16x8 hi = *(const bf16x8*)(p + 16);
  bf16x16 r;
#pragma unroll
  for (int i = 0; i < 8; ++i) { r[i] = lo[i]; r[i + 8] = hi[i]; }
  return r;
}

// B-fragment (32x16, bf16). Source stored row-major per OUTPUT column:
// B[k][n] = base[(row0+n)*stride + k0 + k].
// lane l: n = l&15 ; element e -> k = (l>=16?8:0) + (e&7) + (e>=8?16:0)
__device__ __forceinline__ bf16x16 load_b_frag(const __bf16* base, int stride,
                                               int row0, int k0) {
  int l  = threadIdx.x & 31;
  int n  = l & 15;
  int kh = (l & 16) ? 8 : 0;
  const __bf16* p = base + (size_t)(row0 + n) * stride + (k0 + kh);
  bf16x8 lo = *(const bf16x8*)(p);
  bf16x8 hi = *(const bf16x8*)(p + 16);
  bf16x16 r;
#pragma unroll
  for (int i = 0; i < 8; ++i) { r[i] = lo[i]; r[i + 8] = hi[i]; }
  return r;
}

// ------------------------------------------------- async global->LDS staging
__device__ __forceinline__ void async_ld_b128(const __bf16* g, __bf16* l) {
#if __has_builtin(__builtin_amdgcn_global_load_async_to_lds_b128)
  __builtin_amdgcn_global_load_async_to_lds_b128(
      (__attribute__((address_space(1))) v4i_vs*)g,
      (__attribute__((address_space(3))) v4i_vs*)l, 0, 0);
#else
  *(bf16x8*)l = *(const bf16x8*)g;   // synchronous fallback (reg staged)
#endif
}

__device__ __forceinline__ void wait_async0() {
#if __has_builtin(__builtin_amdgcn_global_load_async_to_lds_b128)
#if __has_builtin(__builtin_amdgcn_s_wait_asynccnt)
  __builtin_amdgcn_s_wait_asynccnt(0);
#else
  asm volatile("s_wait_asynccnt 0" ::: "memory");
#endif
#endif
}

// ---------------------------------------------------------------- small kernels
__global__ void gate_sigmoid_kernel(const float* __restrict__ alpha,
                                    const int* __restrict__ label,
                                    float* __restrict__ sig) {
  int i = blockIdx.x * blockDim.x + threadIdx.x;
  if (i >= Bn * D3n) return;
  int b = i / D3n, d = i % D3n;
  float a = alpha[(size_t)label[b] * D3n + d];
  sig[i] = 1.0f / (1.0f + __expf(-a));
}

__global__ void cvt_f32_bf16_kernel(const float* __restrict__ in,
                                    __bf16* __restrict__ out, int n) {
  int i = blockIdx.x * blockDim.x + threadIdx.x;
  if (i < n) out[i] = (__bf16)in[i];
}

// ---------------------------------------------------------------- gated QKV GEMM
// one wave = 16 tokens x 64 features. qkv = (x @ Wqkv^T + b)*sig, scattered to
// Q (B,H,N,64), K (B,H,N,64), V^T (B,H,64,N), all bf16.
__global__ void __launch_bounds__(256)
qkv_gemm_kernel(const __bf16* __restrict__ xbf, const __bf16* __restrict__ wq,
                const float* __restrict__ qkv_b, const float* __restrict__ sig,
                __bf16* __restrict__ qbuf, __bf16* __restrict__ kbuf,
                __bf16* __restrict__ vT) {
  int wave   = blockIdx.x * 8 + (threadIdx.x >> 5);
  int strip  = wave % 36;            // 36 strips of 64 features over 3C=2304
  int ttile  = wave / 36;            // 1024 tiles of 16 tokens
  int token0 = ttile * 16;
  int d0     = strip * 64;
  int lane   = threadIdx.x & 31;
  int n      = lane & 15;
  int half8  = (lane & 16) ? 8 : 0;

  f32x8 acc[4];
#pragma unroll
  for (int j = 0; j < 4; ++j)
#pragma unroll
    for (int v = 0; v < 8; ++v) acc[j][v] = 0.0f;

  for (int k0 = 0; k0 < Cn; k0 += 32) {
    bf16x16 a = load_a_frag(xbf, Cn, token0, k0);
#pragma unroll
    for (int j = 0; j < 4; ++j) {
      bf16x16 b = load_b_frag(wq, Cn, d0 + j * 16, k0);
      acc[j] = wmma_bf16(a, b, acc[j]);
    }
  }

  int bidx = token0 >> 10;          // batch index (N=1024 tokens each)
  int sect = d0 / Cn;               // 0=q 1=k 2=v
  int dl0  = d0 - sect * Cn;        // multiple of 64 -> exactly one head
  int h    = dl0 / HDn;
  size_t headrow = ((size_t)bidx * Hn + h) * Nn;

#pragma unroll
  for (int j = 0; j < 4; ++j) {
    int d     = d0 + j * 16 + n;
    float g   = sig[(size_t)bidx * D3n + d];
    float bia = qkv_b[d] * g;
    int hd    = j * 16 + n;
#pragma unroll
    for (int v = 0; v < 8; ++v) {
      int tok = (token0 & (Nn - 1)) + v + half8;
      __bf16 val = (__bf16)(acc[j][v] * g + bia);
      if (sect == 0)
        qbuf[(headrow + tok) * HDn + hd] = val;
      else if (sect == 1)
        kbuf[(headrow + tok) * HDn + hd] = val;
      else
        vT[(((size_t)bidx * Hn + h) * HDn + hd) * Nn + tok] = val;
    }
  }
}

// ---------------------------------------------------------------- flash attention
// block = 8 waves, all on the same (b,h); each wave owns one 16-query tile.
// Per 32-key chunk: K (32x64) and V^T (64x32) tiles staged once into LDS via
// async-to-LDS, then consumed by all 8 waves. Online softmax in C-fragment
// layout; P transposed to A-fragment layout through a wave-private LDS region.
__global__ void __launch_bounds__(256)
attn_kernel(const __bf16* __restrict__ qbuf, const __bf16* __restrict__ kbuf,
            const __bf16* __restrict__ vT, __bf16* __restrict__ attn_out) {
  __shared__ __align__(16) __bf16 Ks[32 * 64];        // keys x hd
  __shared__ __align__(16) __bf16 Vs[64 * 32];        // hd x keys
  __shared__ __align__(32) __bf16 psm[8 * 16 * 32];   // per-wave P transpose
  int tid   = threadIdx.x;
  int wid   = tid >> 5;
  int lane  = tid & 31;
  int bh    = blockIdx.x >> 3;
  int q0    = (blockIdx.x & 7) * 128 + wid * 16;
  int n     = lane & 15;
  int half8 = (lane & 16) ? 8 : 0;

  const __bf16* qh = qbuf + (size_t)bh * Nn * HDn;
  const __bf16* kh = kbuf + (size_t)bh * Nn * HDn;
  const __bf16* vh = vT + (size_t)bh * HDn * Nn;
  __bf16* sm = psm + wid * (16 * 32);

  bf16x16 aq0 = load_a_frag(qh, HDn, q0, 0);
  bf16x16 aq1 = load_a_frag(qh, HDn, q0, 32);

  float mrow[8], lrow[8], corr[8];
  f32x8 o[4];
#pragma unroll
  for (int v = 0; v < 8; ++v) { mrow[v] = -3.0e38f; lrow[v] = 0.0f; }
#pragma unroll
  for (int j = 0; j < 4; ++j)
#pragma unroll
    for (int v = 0; v < 8; ++v) o[j][v] = 0.0f;

  const float scale = 0.125f;  // hd^-0.5, hd=64

  for (int t = 0; t < Nn; t += 32) {
    // ---- stage K chunk (32 rows x 128B) + V^T chunk (64 rows x 64B) in LDS
    // 256 threads: one b128 each per matrix.
    async_ld_b128(kh + (size_t)(t + (tid >> 3)) * HDn + (tid & 7) * 8,
                  Ks + tid * 8);
    async_ld_b128(vh + (size_t)(tid >> 2) * Nn + t + (tid & 3) * 8,
                  Vs + tid * 8);
    wait_async0();
    __syncthreads();

    int tn = (t + 32 < Nn) ? t + 32 : t;
    __builtin_prefetch(kh + (size_t)tn * HDn, 0, 1);   // global_prefetch_b8
    __builtin_prefetch(vh + tn, 0, 1);

    f32x8 s0, s1;
#pragma unroll
    for (int v = 0; v < 8; ++v) { s0[v] = 0.0f; s1[v] = 0.0f; }
    // S = Q K^T from LDS (contraction over hd=64 -> two k-chunks per key tile)
    s0 = wmma_bf16(aq0, load_b_frag(Ks, HDn, 0, 0), s0);
    s0 = wmma_bf16(aq1, load_b_frag(Ks, HDn, 0, 32), s0);
    s1 = wmma_bf16(aq0, load_b_frag(Ks, HDn, 16, 0), s1);
    s1 = wmma_bf16(aq1, load_b_frag(Ks, HDn, 16, 32), s1);

#pragma unroll
    for (int v = 0; v < 8; ++v) {
      float x0 = s0[v] * scale, x1 = s1[v] * scale;
      float mx = fmaxf(x0, x1);
      mx = fmaxf(mx, __shfl_xor(mx, 1, 32));
      mx = fmaxf(mx, __shfl_xor(mx, 2, 32));
      mx = fmaxf(mx, __shfl_xor(mx, 4, 32));
      mx = fmaxf(mx, __shfl_xor(mx, 8, 32));
      float mnew = fmaxf(mrow[v], mx);
      float c = __expf(mrow[v] - mnew);
      mrow[v] = mnew;
      float p0 = __expf(x0 - mnew);
      float p1 = __expf(x1 - mnew);
      float rs = p0 + p1;
      rs += __shfl_xor(rs, 1, 32);
      rs += __shfl_xor(rs, 2, 32);
      rs += __shfl_xor(rs, 4, 32);
      rs += __shfl_xor(rs, 8, 32);
      lrow[v] = lrow[v] * c + rs;
      corr[v] = c;
      // transpose P into A-frag layout via wave-private LDS (DS ops in-order)
      sm[(v + half8) * 32 + n]      = (__bf16)p0;
      sm[(v + half8) * 32 + 16 + n] = (__bf16)p1;
    }
#pragma unroll
    for (int j = 0; j < 4; ++j)
#pragma unroll
      for (int v = 0; v < 8; ++v) o[j][v] *= corr[v];

    bf16x16 pa = load_a_frag(sm, 32, 0, 0);           // P as A-frag (16x32)
#pragma unroll
    for (int j = 0; j < 4; ++j) {
      bf16x16 bv = load_b_frag(Vs, 32, j * 16, 0);    // V^T rows = hd, k = key
      o[j] = wmma_bf16(pa, bv, o[j]);
    }
    __syncthreads();   // protect Ks/Vs before next chunk overwrites
  }

  int b = bh / Hn, h = bh % Hn;
#pragma unroll
  for (int v = 0; v < 8; ++v) {
    float inv = 1.0f / lrow[v];
    int tok = q0 + v + half8;
    size_t rowoff = ((size_t)b * Nn + tok) * Cn + h * HDn;
#pragma unroll
    for (int j = 0; j < 4; ++j)
      attn_out[rowoff + j * 16 + n] = (__bf16)(o[j][v] * inv);
  }
}

// ---------------------------------------------------------------- projection GEMM
__global__ void __launch_bounds__(256)
proj_gemm_kernel(const __bf16* __restrict__ abf, const __bf16* __restrict__ wp,
                 const float* __restrict__ pb, float* __restrict__ out) {
  int wave   = blockIdx.x * 8 + (threadIdx.x >> 5);
  int strip  = wave % 12;            // 12 strips of 64 over C=768
  int ttile  = wave / 12;            // 1024 token tiles
  int token0 = ttile * 16;
  int c0     = strip * 64;
  int lane   = threadIdx.x & 31;
  int n      = lane & 15;
  int half8  = (lane & 16) ? 8 : 0;

  f32x8 acc[4];
#pragma unroll
  for (int j = 0; j < 4; ++j)
#pragma unroll
    for (int v = 0; v < 8; ++v) acc[j][v] = 0.0f;

  for (int k0 = 0; k0 < Cn; k0 += 32) {
    bf16x16 a = load_a_frag(abf, Cn, token0, k0);
#pragma unroll
    for (int j = 0; j < 4; ++j) {
      bf16x16 b = load_b_frag(wp, Cn, c0 + j * 16, k0);
      acc[j] = wmma_bf16(a, b, acc[j]);
    }
  }

#pragma unroll
  for (int j = 0; j < 4; ++j) {
    float bias = pb[c0 + j * 16 + n];
#pragma unroll
    for (int v = 0; v < 8; ++v)
      out[(size_t)(token0 + v + half8) * Cn + c0 + j * 16 + n] =
          acc[j][v] + bias;
  }
}

// ---------------------------------------------------------------- launcher
extern "C" void kernel_launch(void* const* d_in, const int* in_sizes, int n_in,
                              void* d_out, int out_size, void* d_ws,
                              size_t ws_size, hipStream_t stream) {
  (void)in_sizes; (void)n_in; (void)out_size; (void)ws_size;
  const float* x      = (const float*)d_in[0];
  const int*   label  = (const int*)d_in[1];
  const float* alpha  = (const float*)d_in[2];
  const float* qkv_w  = (const float*)d_in[3];
  const float* qkv_b  = (const float*)d_in[4];
  const float* proj_w = (const float*)d_in[5];
  const float* proj_b = (const float*)d_in[6];
  float* out = (float*)d_out;

  char* ws = (char*)d_ws;
  size_t off = 0;
  auto carve = [&](size_t bytes) -> void* {
    void* p = ws + off;
    off = (off + bytes + 255) & ~(size_t)255;
    return p;
  };
  float*  sig   = (float*)carve((size_t)Bn * D3n * 4);
  __bf16* xa_bf = (__bf16*)carve((size_t)Bn * Nn * Cn * 2);  // x bf16; reused as attn out
  __bf16* wq_bf = (__bf16*)carve((size_t)D3n * Cn * 2);
  __bf16* wp_bf = (__bf16*)carve((size_t)Cn * Cn * 2);
  __bf16* qbuf  = (__bf16*)carve((size_t)Bn * Hn * Nn * HDn * 2);
  __bf16* kbuf  = (__bf16*)carve((size_t)Bn * Hn * Nn * HDn * 2);
  __bf16* vT    = (__bf16*)carve((size_t)Bn * Hn * HDn * Nn * 2);

  gate_sigmoid_kernel<<<(Bn * D3n) / 256, 256, 0, stream>>>(alpha, label, sig);

  int nx = Bn * Nn * Cn;
  cvt_f32_bf16_kernel<<<(nx + 255) / 256, 256, 0, stream>>>(x, xa_bf, nx);
  int nwq = D3n * Cn;
  cvt_f32_bf16_kernel<<<(nwq + 255) / 256, 256, 0, stream>>>(qkv_w, wq_bf, nwq);
  int nwp = Cn * Cn;
  cvt_f32_bf16_kernel<<<(nwp + 255) / 256, 256, 0, stream>>>(proj_w, wp_bf, nwp);

  // 1024 token tiles * 36 feature strips / 8 waves per block
  qkv_gemm_kernel<<<(1024 * 36) / 8, 256, 0, stream>>>(xa_bf, wq_bf, qkv_b, sig,
                                                       qbuf, kbuf, vT);

  // B*H head-slices * 8 query blocks of 128 (reuses xa_bf as bf16 attn output)
  attn_kernel<<<Bn * Hn * 8, 256, 0, stream>>>(qbuf, kbuf, vT, xa_bf);

  // 1024 token tiles * 12 feature strips / 8 waves per block
  proj_gemm_kernel<<<(1024 * 12) / 8, 256, 0, stream>>>(xa_bf, wp_bf, proj_b,
                                                        out);
}
